// GatedAttnStudyHead_87591563035146
// MI455X (gfx1250) — matrix operand order repaired and checked by reading, not compiled, not run on hardware
//
#include <hip/hip_runtime.h>
#include <hip/hip_bf16.h>
#include <math.h>

typedef __bf16 bf16_t;
typedef __attribute__((ext_vector_type(16))) __bf16 v16bf;
typedef __attribute__((ext_vector_type(8)))  __bf16 v8bf;
typedef __attribute__((ext_vector_type(8)))  float  v8f;

#define EMB 512
#define ATTN 256
#define HID 512
#define NCLS 14

// ---------------- workspace layout (bytes) ----------------
#define OFF_BV    (0u)                      // 131072 bf16 = 262144 B
#define OFF_BU    (262144u)                 // 131072 bf16
#define OFF_W1P   (524288u)                 // 262144 bf16 = 524288 B
#define OFF_W2P   (1048576u)                // 8192 bf16 = 16384 B
#define OFF_A     (1064960u)                // N f32
#define OFF_E     (2703360u)                // N f32
#define OFF_SMAX  (4341760u)                // B f32
#define OFF_DEN   (4358144u)                // B f32
#define OFF_Z     (4374528u)                // B*512 f32

static __device__ __forceinline__ v8f wmma_bf16(v16bf a, v16bf b, v8f c) {
  return __builtin_amdgcn_wmma_f32_16x16x32_bf16(false, a, false, b, (short)0, c, false, false);
}

// Async DMA: 16 bytes memory -> LDS, tracked by ASYNCcnt (CDNA5 §10.7).
static __device__ __forceinline__ void async_b128(unsigned lds_byte_off,
                                                  const void* sbase,
                                                  unsigned byte_off) {
  asm volatile("global_load_async_to_lds_b128 %0, %1, %2"
               :: "v"(lds_byte_off), "v"(byte_off), "s"(sbase)
               : "memory");
}
static __device__ __forceinline__ void wait_async0() {
  asm volatile("s_wait_asynccnt 0" ::: "memory");
}

// ---------------------------------------------------------------------------
// Pack weight [K x ncols_src] (row major) into WMMA B-fragment order:
// element index = ((nt*16 + kt)*32 + lane)*16 + j
// with k = kt*32 + (lane>=16 ? 16:0) + j, n = nt*16 + (lane&15).
// Pads n >= ncols_src with zeros (W2: 14 -> 16).
// ---------------------------------------------------------------------------
__global__ void pack_b_kernel(const float* __restrict__ W, bf16_t* __restrict__ dst,
                              int ncols_src, int total) {
  int idx = blockIdx.x * blockDim.x + threadIdx.x;
  if (idx >= total) return;
  int j    = idx & 15;
  int lane = (idx >> 4) & 31;
  int kt   = (idx >> 9) & 15;
  int nt   = idx >> 13;
  int k = kt * 32 + ((lane >= 16) ? 16 : 0) + j;
  int n = nt * 16 + (lane & 15);
  float v = (n < ncols_src) ? W[(size_t)k * ncols_src + n] : 0.0f;
  dst[idx] = (bf16_t)v;
}

__global__ void init_kernel(float* smax, float* denom, float* z, int B) {
  int i = blockIdx.x * blockDim.x + threadIdx.x;
  if (i < B) { smax[i] = -INFINITY; denom[i] = 0.0f; }
  int total = B * EMB;
  if (i < total) z[i] = 0.0f;
}

// ---------------------------------------------------------------------------
// Gated attention score: a[i] = ww . (tanh(xV+Vb) * sigmoid(xU+Ub)) + wb
// One wave = 16 instances, 4 waves/block.  B fragments are staged once per
// block into LDS with async-to-LDS DMA, double buffered across the N loop.
// LDS: 64KB X-staging buffer reused as 2 x 32KB B double buffer.
// ---------------------------------------------------------------------------
#define SC_WAVES 4
__global__ void __launch_bounds__(SC_WAVES * 32)
score_kernel(const float* __restrict__ X,
             const bf16_t* __restrict__ bv, const bf16_t* __restrict__ bu,
             const float* __restrict__ Vb, const float* __restrict__ Ub,
             const float* __restrict__ ww, const float* __restrict__ wbp,
             float* __restrict__ a_out, int N) {
  __shared__ __align__(128) unsigned char sraw[65536];

  const int tid  = threadIdx.x;
  const int lane = tid & 31;
  const int wv   = tid >> 5;
  int row0 = (blockIdx.x * SC_WAVES + wv) * 16;
  if (row0 + 16 > N) row0 = N - 16;  // clamp (keep barriers uniform)

  // ---- stage own 16x512 f32 tile -> bf16 LDS (coalesced) ----
  {
    const float* xt = X + (size_t)row0 * EMB;
    bf16_t* dst = (bf16_t*)sraw + (size_t)wv * 16 * EMB;
    #pragma unroll
    for (int t = lane * 4; t < 16 * EMB; t += 32 * 4) {
      float4 f = *(const float4*)(xt + t);
      dst[t + 0] = (bf16_t)f.x;
      dst[t + 1] = (bf16_t)f.y;
      dst[t + 2] = (bf16_t)f.z;
      dst[t + 3] = (bf16_t)f.w;
    }
  }
  __syncthreads();

  // ---- build 16 A fragments (K chunks of 32) ----
  const int arow = lane & 15;
  const int kh   = (lane < 16) ? 0 : 16;
  v16bf afrag[16];
  {
    const bf16_t* xr = (const bf16_t*)sraw + (size_t)wv * 16 * EMB + (size_t)arow * EMB;
    #pragma unroll
    for (int kt = 0; kt < 16; ++kt) {
      const bf16_t* p = xr + kt * 32 + kh;
      v8bf lo = *(const v8bf*)(p);
      v8bf hi = *(const v8bf*)(p + 8);
      afrag[kt] = __builtin_shufflevector(lo, hi, 0,1,2,3,4,5,6,7,8,9,10,11,12,13,14,15);
    }
  }
  __syncthreads();  // all waves done reading X; sraw becomes B double buffer

  bf16_t* bbuf0 = (bf16_t*)sraw;            // [Bv 16KB][Bu 16KB]
  bf16_t* bbuf1 = (bf16_t*)(sraw + 32768);

  // prologue: async stage nt=0 B tiles (32KB / 128 threads = 16 x b128 each)
  {
    unsigned lbase = (unsigned)(uintptr_t)bbuf0;
    #pragma unroll
    for (int i = 0; i < 8; ++i) {
      unsigned off = (unsigned)((tid + i * 128) * 16);
      async_b128(lbase + off,         (const void*)bv, off);
      async_b128(lbase + 16384 + off, (const void*)bu, off);
    }
  }
  wait_async0();
  __syncthreads();

  float apart[8];
  #pragma unroll
  for (int r = 0; r < 8; ++r) apart[r] = 0.0f;

  #pragma unroll 1
  for (int nt = 0; nt < ATTN / 16; ++nt) {
    bf16_t* cur = (nt & 1) ? bbuf1 : bbuf0;
    bf16_t* nxt = (nt & 1) ? bbuf0 : bbuf1;

    // kick off DMA of next tile; overlaps the 32 WMMAs below
    if (nt < ATTN / 16 - 1) {
      const bf16_t* gv = bv + (size_t)(nt + 1) * 8192;
      const bf16_t* gu = bu + (size_t)(nt + 1) * 8192;
      unsigned lbase = (unsigned)(uintptr_t)nxt;
      #pragma unroll
      for (int i = 0; i < 8; ++i) {
        unsigned off = (unsigned)((tid + i * 128) * 16);
        async_b128(lbase + off,         (const void*)gv, off);
        async_b128(lbase + 16384 + off, (const void*)gu, off);
      }
    }

    v8f accv = {};
    v8f accu = {};
    const bf16_t* bvp = cur + (size_t)lane * 16;
    const bf16_t* bup = cur + 8192 + (size_t)lane * 16;
    #pragma unroll
    for (int kt = 0; kt < 16; ++kt) {
      v16bf bfv = *(const v16bf*)(bvp + (size_t)kt * 512);
      v16bf bfu = *(const v16bf*)(bup + (size_t)kt * 512);
      accv = wmma_bf16(afrag[kt], bfv, accv);
      accu = wmma_bf16(afrag[kt], bfu, accu);
    }

    const int n = nt * 16 + (lane & 15);
    const float vbn = Vb[n], ubn = Ub[n], wn = ww[n];
    #pragma unroll
    for (int r = 0; r < 8; ++r) {
      float t = tanhf(accv[r] + vbn);
      float s = 1.0f / (1.0f + __expf(-(accu[r] + ubn)));
      apart[r] += t * s * wn;
    }

    wait_async0();
    __syncthreads();
  }

  // ---- reduce over the 16 columns held by each half-wave ----
  #pragma unroll
  for (int r = 0; r < 8; ++r) {
    float v = apart[r];
    v += __shfl_xor(v, 8, 16);
    v += __shfl_xor(v, 4, 16);
    v += __shfl_xor(v, 2, 16);
    v += __shfl_xor(v, 1, 16);
    apart[r] = v;
  }
  if ((lane & 15) == 0) {
    const float wb0 = wbp[0];
    const int mb = (lane < 16) ? 0 : 8;
    #pragma unroll
    for (int r = 0; r < 8; ++r) a_out[row0 + mb + r] = apart[r] + wb0;
  }
}

// ---------------------------------------------------------------------------
__global__ void segmax_kernel(const float* __restrict__ a, const int* __restrict__ g,
                              float* __restrict__ smax, int N) {
  int i = blockIdx.x * blockDim.x + threadIdx.x;
  if (i >= N) return;
  float v = a[i];
  int* p = (int*)&smax[g[i]];
  if (v >= 0.0f) atomicMax(p, __float_as_int(v));
  else           atomicMin((unsigned int*)p, (unsigned int)__float_as_int(v));
}

__global__ void expsum_kernel(const float* __restrict__ a, const int* __restrict__ g,
                              const float* __restrict__ smax, float* __restrict__ e,
                              float* __restrict__ denom, int N) {
  int i = blockIdx.x * blockDim.x + threadIdx.x;
  if (i >= N) return;
  int gi = g[i];
  float m = smax[gi];
  if (m == -INFINITY) m = 0.0f;
  float ev = __expf(a[i] - m);
  e[i] = ev;
  atomicAdd(&denom[gi], ev);
}

__global__ void scatter_kernel(const float* __restrict__ X, const float* __restrict__ e,
                               const float* __restrict__ denom, const int* __restrict__ g,
                               float* __restrict__ z, int N) {
  int inst = blockIdx.x * (blockDim.x >> 5) + (threadIdx.x >> 5);
  if (inst >= N) return;
  int lane = threadIdx.x & 31;
  int gi = g[inst];
  float d = denom[gi];
  float attn = e[inst] / (d > 0.0f ? d : 1.0f);
  const float* xr = X + (size_t)inst * EMB;
  float* zr = z + (size_t)gi * EMB;
  #pragma unroll
  for (int c = lane; c < EMB; c += 32) atomicAdd(&zr[c], attn * xr[c]);
}

// ---------------------------------------------------------------------------
// Classifier: logits = relu(z@W1 + b1) @ W2 + b2.  One wave = 16 rows.
// ---------------------------------------------------------------------------
#define CL_WAVES 2
__global__ void __launch_bounds__(CL_WAVES * 32)
classify_kernel(const float* __restrict__ z,
                const bf16_t* __restrict__ w1p, const bf16_t* __restrict__ w2p,
                const float* __restrict__ b1, const float* __restrict__ b2,
                float* __restrict__ out, int B) {
  __shared__ bf16_t zsm[CL_WAVES][16 * EMB];
  __shared__ bf16_t hsm[CL_WAVES][16 * HID];

  const int lane = threadIdx.x & 31;
  const int wv   = threadIdx.x >> 5;
  int row0 = (blockIdx.x * CL_WAVES + wv) * 16;
  if (row0 + 16 > B) row0 = B - 16;

  {
    const float* zt = z + (size_t)row0 * EMB;
    bf16_t* dst = &zsm[wv][0];
    #pragma unroll
    for (int t = lane * 4; t < 16 * EMB; t += 32 * 4) {
      float4 f = *(const float4*)(zt + t);
      dst[t + 0] = (bf16_t)f.x;
      dst[t + 1] = (bf16_t)f.y;
      dst[t + 2] = (bf16_t)f.z;
      dst[t + 3] = (bf16_t)f.w;
    }
  }
  __syncthreads();

  const int arow = lane & 15;
  const int kh   = (lane < 16) ? 0 : 16;
  const int mb   = (lane < 16) ? 0 : 8;

  v16bf afrag[16];
  {
    const bf16_t* xr = &zsm[wv][0] + (size_t)arow * EMB;
    #pragma unroll
    for (int kt = 0; kt < 16; ++kt) {
      const bf16_t* p = xr + kt * 32 + kh;
      v8bf lo = *(const v8bf*)(p);
      v8bf hi = *(const v8bf*)(p + 8);
      afrag[kt] = __builtin_shufflevector(lo, hi, 0,1,2,3,4,5,6,7,8,9,10,11,12,13,14,15);
    }
  }

  #pragma unroll 1
  for (int nt = 0; nt < HID / 16; ++nt) {
    v8f acc = {};
    const bf16_t* wp = w1p + ((size_t)nt * 16 * 32 + lane) * 16;
    #pragma unroll
    for (int kt = 0; kt < 16; ++kt) {
      v16bf bf = *(const v16bf*)(wp + (size_t)kt * 512);
      acc = wmma_bf16(afrag[kt], bf, acc);
    }
    const int n = nt * 16 + (lane & 15);
    const float b1n = b1[n];
    #pragma unroll
    for (int r = 0; r < 8; ++r) {
      hsm[wv][(mb + r) * HID + n] = (bf16_t)fmaxf(acc[r] + b1n, 0.0f);
    }
  }
  __syncthreads();

  {
    const bf16_t* xr = &hsm[wv][0] + (size_t)arow * HID;
    #pragma unroll
    for (int kt = 0; kt < 16; ++kt) {
      const bf16_t* p = xr + kt * 32 + kh;
      v8bf lo = *(const v8bf*)(p);
      v8bf hi = *(const v8bf*)(p + 8);
      afrag[kt] = __builtin_shufflevector(lo, hi, 0,1,2,3,4,5,6,7,8,9,10,11,12,13,14,15);
    }
  }

  v8f acc = {};
  {
    const bf16_t* wp = w2p + (size_t)lane * 16;
    #pragma unroll
    for (int kt = 0; kt < 16; ++kt) {
      v16bf bf = *(const v16bf*)(wp + (size_t)kt * 512);
      acc = wmma_bf16(afrag[kt], bf, acc);
    }
  }
  const int n = lane & 15;
  if (n < NCLS) {
    const float b2n = b2[n];
    #pragma unroll
    for (int r = 0; r < 8; ++r) {
      out[(size_t)(row0 + mb + r) * NCLS + n] = acc[r] + b2n;
    }
  }
}

// ---------------------------------------------------------------------------
extern "C" void kernel_launch(void* const* d_in, const int* in_sizes, int n_in,
                              void* d_out, int out_size, void* d_ws, size_t ws_size,
                              hipStream_t stream) {
  (void)n_in; (void)ws_size;
  const float* X    = (const float*)d_in[0];
  const int*   gidx = (const int*)d_in[1];
  const float* Vw = (const float*)d_in[4];
  const float* Vb = (const float*)d_in[5];
  const float* Uw = (const float*)d_in[6];
  const float* Ub = (const float*)d_in[7];
  const float* ww = (const float*)d_in[8];
  const float* wb = (const float*)d_in[9];
  const float* W1 = (const float*)d_in[10];
  const float* b1 = (const float*)d_in[11];
  const float* W2 = (const float*)d_in[12];
  const float* b2 = (const float*)d_in[13];
  float* out = (float*)d_out;

  const int N = in_sizes[1];       // instances
  const int B = out_size / NCLS;   // studies

  char* ws = (char*)d_ws;
  bf16_t* bv   = (bf16_t*)(ws + OFF_BV);
  bf16_t* bu   = (bf16_t*)(ws + OFF_BU);
  bf16_t* w1p  = (bf16_t*)(ws + OFF_W1P);
  bf16_t* w2p  = (bf16_t*)(ws + OFF_W2P);
  float*  a    = (float*)(ws + OFF_A);
  float*  e    = (float*)(ws + OFF_E);
  float*  smax = (float*)(ws + OFF_SMAX);
  float*  den  = (float*)(ws + OFF_DEN);
  float*  z    = (float*)(ws + OFF_Z);

  pack_b_kernel<<<(131072 + 255) / 256, 256, 0, stream>>>(Vw, bv, ATTN, 131072);
  pack_b_kernel<<<(131072 + 255) / 256, 256, 0, stream>>>(Uw, bu, ATTN, 131072);
  pack_b_kernel<<<(262144 + 255) / 256, 256, 0, stream>>>(W1, w1p, HID, 262144);
  pack_b_kernel<<<(8192 + 255) / 256, 256, 0, stream>>>(W2, w2p, NCLS, 8192);

  init_kernel<<<(B * EMB + 255) / 256, 256, 0, stream>>>(smax, den, z, B);

  {
    int tiles = (N + 15) / 16;
    int blocks = (tiles + SC_WAVES - 1) / SC_WAVES;
    score_kernel<<<blocks, SC_WAVES * 32, 0, stream>>>(X, bv, bu, Vb, Ub, ww, wb, a, N);
  }

  segmax_kernel<<<(N + 255) / 256, 256, 0, stream>>>(a, gidx, smax, N);
  expsum_kernel<<<(N + 255) / 256, 256, 0, stream>>>(a, gidx, smax, e, den, N);
  scatter_kernel<<<(N + 7) / 8, 256, 0, stream>>>(X, e, den, gidx, z, N);

  {
    int tiles = (B + 15) / 16;
    int blocks = (tiles + CL_WAVES - 1) / CL_WAVES;
    classify_kernel<<<blocks, CL_WAVES * 32, 0, stream>>>(z, w1p, w2p, b1, b2, out, B);
  }
}
